// LSTM_Layer_83099027243422
// MI455X (gfx1250) — matrix operand order repaired and checked by reading, not compiled, not run on hardware
//
#include <hip/hip_runtime.h>

// Problem constants
#define BB   32
#define TT   1024
#define II   512
#define HH   512
#define GG   2048           // 4*H
#define BT   (BB*TT)        // 32768

#define LDSTR_H  40         // padded row stride (halves) for h LDS (80B, 16B aligned)
#define LDSTR_W  72         // padded row stride (halves) for W LDS (144B, 16B aligned)

typedef __attribute__((ext_vector_type(16))) __bf16 v16bf;
typedef __attribute__((ext_vector_type(8)))  float  v8f;

union Frag16 { v16bf v; unsigned short h[16]; uint4 q[2]; };
union Acc    { v8f v; float f[8]; };
union H4     { uint2 u; unsigned short h[4]; };

__device__ __forceinline__ unsigned short f32_to_bf16(float f) {
    unsigned u = __builtin_bit_cast(unsigned, f);
    unsigned r = (u + 0x7FFFu + ((u >> 16) & 1u)) >> 16;   // round-to-nearest-even
    return (unsigned short)r;
}
__device__ __forceinline__ float bf16_to_f32(unsigned short s) {
    unsigned u = ((unsigned)s) << 16;
    return __builtin_bit_cast(float, u);
}
__device__ __forceinline__ float sigf(float x) { return 1.0f / (1.0f + __expf(-x)); }

// gfx1250 async global->LDS copy (16B per lane), tracked by ASYNCcnt
__device__ __forceinline__ void async_copy_b128(const void* gsrc, void* lds_dst) {
    unsigned ldsOff = (unsigned)(size_t)lds_dst;   // LDS aperture: addr[31:0] == LDS offset
    asm volatile("global_load_async_to_lds_b128 %0, %1, off"
                 :: "v"(ldsOff), "v"(gsrc) : "memory");
}
__device__ __forceinline__ void async_wait0() {
    asm volatile("s_wait_asynccnt 0x0" ::: "memory");
}
__device__ __forceinline__ void async_wait1() {   // allow 1 copy in flight (pipelining)
    asm volatile("s_wait_asynccnt 0x1" ::: "memory");
}

// ---------------------------------------------------------------------------
// f32 -> bf16 conversion (grid-stride)
// ---------------------------------------------------------------------------
__global__ void cvt_bf16_kernel(const float* __restrict__ src,
                                unsigned short* __restrict__ dst, int n) {
    int i = blockIdx.x * blockDim.x + threadIdx.x;
    int stride = gridDim.x * blockDim.x;
    for (; i < n; i += stride) dst[i] = f32_to_bf16(src[i]);
}

// f32 [rows][cols] -> bf16 transposed [cols][rows]
__global__ void cvt_bf16_T_kernel(const float* __restrict__ src,
                                  unsigned short* __restrict__ dst,
                                  int rows, int cols) {
    int i = blockIdx.x * blockDim.x + threadIdx.x;
    int stride = gridDim.x * blockDim.x;
    int n = rows * cols;
    for (; i < n; i += stride) {
        int r = i / cols, c = i % cols;            // coalesced read over c
        dst[(size_t)c * rows + r] = f32_to_bf16(src[i]);
    }
}

// zero-init (uint words)
__global__ void zero_kernel(unsigned* __restrict__ p, int nwords) {
    int i = blockIdx.x * blockDim.x + threadIdx.x;
    int stride = gridDim.x * blockDim.x;
    for (; i < nwords; i += stride) p[i] = 0u;
}

// ---------------------------------------------------------------------------
// wx GEMM: wx[m,g] = sum_i x[m,i]*W[g,i] + b_w[g], stored bf16 [BT][G].
// Double-buffered LDS: async copy of tile k+1 overlaps WMMA on tile k.
// WG: 256 thr = 8 waves, tile 128(M) x 64(N); wave tile 32x32 (2x2 WMMA).
// ---------------------------------------------------------------------------
__global__ __launch_bounds__(256)
void wx_gemm_kernel(const unsigned short* __restrict__ xb,
                    const unsigned short* __restrict__ WbT,   // [I][G] k-major
                    const float* __restrict__ b_w,
                    unsigned short* __restrict__ wx) {
    __shared__ unsigned short ldsW[2][32 * LDSTR_W];   // [buf][k][g_local]

    const int wave = threadIdx.x >> 5;
    const int lane = threadIdx.x & 31;
    const int lrow = lane & 15;
    const int lgr  = lane >> 4;
    const int m0   = blockIdx.x * 128 + (wave >> 1) * 32;
    const int gblk = blockIdx.y * 64;
    const int gw   = (wave & 1) * 32;

    const int srow  = threadIdx.x >> 3;            // staging row 0..31
    const int scol8 = (threadIdx.x & 7) * 8;       // staging col chunk

    Acc c[2][2];
#pragma unroll
    for (int a = 0; a < 2; ++a)
#pragma unroll
        for (int b = 0; b < 2; ++b)
#pragma unroll
            for (int r = 0; r < 8; ++r) c[a][b].f[r] = 0.0f;

    // prologue: stage k-step 0 into buffer 0
    async_copy_b128(WbT + (size_t)(srow)*GG + gblk + scol8,
                    &ldsW[0][srow * LDSTR_W + scol8]);

    for (int s = 0; s < 16; ++s) {
        const int cur = s & 1;
        if (s + 1 < 16) {   // prefetch next tile into the other buffer
            async_copy_b128(WbT + (size_t)((s + 1) * 32 + srow) * GG + gblk + scol8,
                            &ldsW[cur ^ 1][srow * LDSTR_W + scol8]);
            async_wait1();  // current tile's copy complete, next stays in flight
        } else {
            async_wait0();
        }
        __syncthreads();

        const int k0 = s * 32;
        // A fragments: two contiguous 16B loads per lane
        Frag16 a[2];
#pragma unroll
        for (int am = 0; am < 2; ++am) {
            const unsigned short* p =
                xb + (size_t)(m0 + am * 16 + lrow) * II + k0 + 8 * lgr;
            a[am].q[0] = *(const uint4*)p;
            a[am].q[1] = *(const uint4*)(p + 16);
        }
        // B fragments: N = lane%16, K per element in table order (matches A)
        Frag16 b[2];
#pragma unroll
        for (int bn = 0; bn < 2; ++bn) {
            const unsigned short* rp = &ldsW[cur][gw + bn * 16 + lrow];
#pragma unroll
            for (int e = 0; e < 16; ++e)
                b[bn].h[e] = rp[((e & 7) + 8 * lgr + 16 * (e >> 3)) * LDSTR_W];
        }
#pragma unroll
        for (int am = 0; am < 2; ++am)
#pragma unroll
            for (int bn = 0; bn < 2; ++bn)
                c[am][bn].v = __builtin_amdgcn_wmma_f32_16x16x32_bf16(
                    false, a[am].v, false, b[bn].v, (short)0, c[am][bn].v,
                    false, false);
        __syncthreads();
    }

    // epilogue: += b_w, convert, store
#pragma unroll
    for (int am = 0; am < 2; ++am) {
#pragma unroll
        for (int bn = 0; bn < 2; ++bn) {
            int g = gblk + gw + bn * 16 + lrow;            // N = lane%16
            float bw = b_w[g];
#pragma unroll
            for (int r = 0; r < 8; ++r) {
                int m = m0 + am * 16 + r + 8 * lgr;        // M = r + 8*(lane/16)
                wx[(size_t)m * GG + g] = f32_to_bf16(c[am][bn].f[r] + bw);
            }
        }
    }
}

// ---------------------------------------------------------------------------
// Persistent recurrent kernel. 32 blocks x 128 threads = 128 wave32.
// Wave w owns PERMUTED gate rows {gg*512 + 4w + jo : gg=0..3, jo=0..3} x all
// 32 batches, so each wave's accumulator holds complete i/f/g/o for 4 hidden
// units: the LSTM cell update is done IN REGISTERS (one shfl_xor half-swap),
// cell state c lives in registers for all 1024 steps, gates never touch
// memory, and only ONE grid barrier per step (h handoff, double-buffered).
// U slice stays resident in VGPRs (16 A-fragments) for the whole scan.
// ---------------------------------------------------------------------------
__device__ __forceinline__ void grid_barrier(unsigned* bar, unsigned target) {
    __syncthreads();
    if (threadIdx.x == 0) {
        __threadfence();                       // release
        atomicAdd(bar, 1u);
        while (__hip_atomic_load(bar, __ATOMIC_RELAXED,
                                 __HIP_MEMORY_SCOPE_AGENT) < target)
            __builtin_amdgcn_s_sleep(2);
        __threadfence();                       // acquire
    }
    __syncthreads();
}

__global__ __launch_bounds__(128)
void lstm_scan_kernel(const unsigned short* __restrict__ Ub,    // [G][H]
                      const unsigned short* __restrict__ wx,    // [BT][G] bf16
                      const float* __restrict__ b_u,
                      unsigned short* __restrict__ h0buf,       // [H][B] bf16
                      unsigned short* __restrict__ h1buf,       // [H][B] bf16
                      unsigned* __restrict__ bar,
                      float* __restrict__ hs,                   // [B][T][H]
                      float* __restrict__ cs) {                 // [B][T][H]
    __shared__ unsigned short ldsHT[HH * LDSTR_H];   // [k][b] padded, 40KB

    const int wave = threadIdx.x >> 5;
    const int lane = threadIdx.x & 31;
    const int lrow = lane & 15;
    const int lgr  = lane >> 4;
    const int w    = blockIdx.x * 4 + wave;    // 0..127
    const int j0   = w * 4;                    // hidden-unit base for this wave

    // Permuted A-row map: m (0..15) -> global gate row
    //   grow(m) = (m>>2)*512 + j0 + (m&3)   (m>>2 selects i/f/g/o group)
    // Preload U A-fragments (128 VGPRs, resident across the whole t-loop)
    const int arow = ((lrow >> 2) * 512) + j0 + (lrow & 3);
    Frag16 aU[16];
#pragma unroll
    for (int s = 0; s < 16; ++s) {
        const unsigned short* p = Ub + (size_t)arow * HH + s * 32 + 8 * lgr;
        aU[s].q[0] = *(const uint4*)p;
        aU[s].q[1] = *(const uint4*)(p + 16);
    }
    // Preload b_u for this lane's 8 C rows: m = r + 8*lgr
    float bu[8];
#pragma unroll
    for (int r = 0; r < 8; ++r) {
        int m = r + 8 * lgr;
        bu[r] = b_u[(m >> 2) * 512 + j0 + (m & 3)];
    }

    // persistent cell state: (2 batch tiles) x (2 hidden units per lane)
    float cst[2][2] = {{0.0f, 0.0f}, {0.0f, 0.0f}};
    const int jb = j0 + (lgr ? 2 : 0);         // the 2 hidden units this lane owns

    for (int t = 0; t < TT; ++t) {
        const unsigned short* h_r = (t & 1) ? h1buf : h0buf;
        unsigned short*       h_w = (t & 1) ? h0buf : h1buf;

        // async-stage h_r (32KB, linear) into padded LDS rows
        for (int i = threadIdx.x; i < (BB * HH) / 8; i += 128) {
            int row  = i >> 2;                 // k
            int col8 = (i & 3) * 8;            // b chunk
            async_copy_b128(h_r + 8 * (size_t)i, &ldsHT[row * LDSTR_H + col8]);
        }
        async_wait0();
        __syncthreads();

        Acc acc[2];
#pragma unroll
        for (int bt = 0; bt < 2; ++bt)
#pragma unroll
            for (int r = 0; r < 8; ++r) acc[bt].f[r] = 0.0f;

#pragma unroll
        for (int s = 0; s < 16; ++s) {
            const unsigned short* rp = &ldsHT[(s * 32) * LDSTR_H + lrow];
            Frag16 b0, b1;
#pragma unroll
            for (int e = 0; e < 16; ++e) {
                const int kk = ((e & 7) + 8 * lgr + 16 * (e >> 3)) * LDSTR_H;
                b0.h[e] = rp[kk];
                b1.h[e] = rp[kk + 16];
            }
            acc[0].v = __builtin_amdgcn_wmma_f32_16x16x32_bf16(
                false, aU[s].v, false, b0.v, (short)0, acc[0].v, false, false);
            acc[1].v = __builtin_amdgcn_wmma_f32_16x16x32_bf16(
                false, aU[s].v, false, b1.v, (short)0, acc[1].v, false, false);
        }
        __syncthreads();   // LDS reuse next step

#pragma unroll
        for (int bt = 0; bt < 2; ++bt) {
            const int b = bt * 16 + lrow;          // batch column (N = lane%16)
            // add wx(t) + b_u: this lane's rows are two 4-half runs
            const unsigned short* wp =
                wx + ((size_t)(b * TT + t) * GG + (size_t)lgr * 1024 + j0);
            H4 w0, w1;
            w0.u = *(const uint2*)wp;              // gate group 2*lgr
            w1.u = *(const uint2*)(wp + 512);      // gate group 2*lgr+1
#pragma unroll
            for (int r = 0; r < 4; ++r) {
                acc[bt].f[r]     += bf16_to_f32(w0.h[r]) + bu[r];
                acc[bt].f[4 + r] += bf16_to_f32(w1.h[r]) + bu[4 + r];
            }
            // half-wave swap: lanes 0-15 hold i,f; lanes 16-31 hold g,o
            float ps0 = lgr ? acc[bt].f[0] : acc[bt].f[2];
            float ps1 = lgr ? acc[bt].f[1] : acc[bt].f[3];
            float ps2 = lgr ? acc[bt].f[4] : acc[bt].f[6];
            float ps3 = lgr ? acc[bt].f[5] : acc[bt].f[7];
            float p0 = __shfl_xor(ps0, 16, 32);
            float p1 = __shfl_xor(ps1, 16, 32);
            float p2 = __shfl_xor(ps2, 16, 32);
            float p3 = __shfl_xor(ps3, 16, 32);
            float iv0 = lgr ? p0 : acc[bt].f[0];
            float iv1 = lgr ? p1 : acc[bt].f[1];
            float fv0 = lgr ? p2 : acc[bt].f[4];
            float fv1 = lgr ? p3 : acc[bt].f[5];
            float gv0 = lgr ? acc[bt].f[2] : p0;
            float gv1 = lgr ? acc[bt].f[3] : p1;
            float ov0 = lgr ? acc[bt].f[6] : p2;
            float ov1 = lgr ? acc[bt].f[7] : p3;
            // LSTM cell update, c in registers
            float c0 = sigf(fv0) * cst[bt][0] + sigf(iv0) * tanhf(gv0);
            float c1 = sigf(fv1) * cst[bt][1] + sigf(iv1) * tanhf(gv1);
            float h0 = sigf(ov0) * tanhf(c0);
            float h1 = sigf(ov1) * tanhf(c1);
            cst[bt][0] = c0;
            cst[bt][1] = c1;
            // transposed hidden state for next step's linear LDS stage
            h_w[(size_t)(jb + 0) * BB + b] = f32_to_bf16(h0);
            h_w[(size_t)(jb + 1) * BB + b] = f32_to_bf16(h1);
            // outputs (aligned float2: lane owns 2 adjacent hidden units)
            size_t o = ((size_t)b * TT + t) * HH + jb;
            float2 hv; hv.x = h0; hv.y = h1;
            float2 cv; cv.x = c0; cv.y = c1;
            *(float2*)(hs + o) = hv;
            *(float2*)(cs + o) = cv;
        }

        grid_barrier(bar, (unsigned)(t + 1) * gridDim.x);
    }
}

// ---------------------------------------------------------------------------
// Host launcher
// ---------------------------------------------------------------------------
extern "C" void kernel_launch(void* const* d_in, const int* in_sizes, int n_in,
                              void* d_out, int out_size, void* d_ws, size_t ws_size,
                              hipStream_t stream) {
    (void)in_sizes; (void)n_in; (void)out_size; (void)ws_size;

    const float* x   = (const float*)d_in[0];   // [B,T,I]
    const float* W   = (const float*)d_in[1];   // [4H,I]
    const float* b_w = (const float*)d_in[2];   // [4H]
    const float* U   = (const float*)d_in[3];   // [4H,H]
    const float* b_u = (const float*)d_in[4];   // [4H]

    float* hs = (float*)d_out;
    float* cs = hs + (size_t)BB * TT * HH;

    // workspace layout (bytes)
    char* ws = (char*)d_ws;
    unsigned short* xb    = (unsigned short*)(ws);                       // 33,554,432
    unsigned short* WbT   = (unsigned short*)(ws + 33554432);            //  2,097,152 [I][G]
    unsigned short* Ub    = (unsigned short*)(ws + 35651584);            //  2,097,152 [G][H]
    unsigned short* wx    = (unsigned short*)(ws + 37748736);            // 134,217,728 bf16
    unsigned short* h0buf = (unsigned short*)(ws + 171966464);           //     32,768
    unsigned short* h1buf = (unsigned short*)(ws + 171999232);           //     32,768
    unsigned*       bar   = (unsigned*)      (ws + 172032000);           //        256

    // 1) convert inputs to bf16 (W transposed to k-major)
    cvt_bf16_kernel<<<1024, 256, 0, stream>>>(x, xb, BB * TT * II);
    cvt_bf16_T_kernel<<<256, 256, 0, stream>>>(W, WbT, GG, II);
    cvt_bf16_kernel<<<256, 256, 0, stream>>>(U, Ub, GG * HH);

    // 2) zero h buffers + barrier counter (contiguous region)
    zero_kernel<<<96, 256, 0, stream>>>((unsigned*)h0buf,
                                        (32768 * 2 + 256) / 4);

    // 3) input projection GEMM (WMMA bf16, double-buffered async staging)
    wx_gemm_kernel<<<dim3(BT / 128, GG / 64), 256, 0, stream>>>(xb, WbT, b_w, wx);

    // 4) persistent recurrent scan (WMMA bf16, in-register cell update,
    //    one grid barrier per step)
    lstm_scan_kernel<<<32, 128, 0, stream>>>(Ub, wx, b_u, h0buf, h1buf,
                                             bar, hs, cs);
}
